// WABlock_25331717112567
// MI455X (gfx1250) — compile-verified, matrix operand order checked by reading, not run on hardware
//
#include <hip/hip_runtime.h>
#include <hip/hip_bf16.h>
#include <math.h>

#define CCH   128          // channels
#define HH    256
#define WW    256
#define WSZ   8
#define NTOK  64           // tokens per window
#define NH    2            // heads
#define DK    64           // head dim
#define NWIN  4096         // 4 * 32 * 32 tiles

typedef __bf16 v16bf __attribute__((ext_vector_type(16)));
typedef float  v8f   __attribute__((ext_vector_type(8)));
typedef int    av4i  __attribute__((vector_size(16)));   // matches builtin's int4 pointee

// ---- workspace layout (bytes) ----
#define WS_BIAS    0                         // 64*64 f32            (16 KB)
#define WS_PKQKV   16384                     // 24*4*32*16 bf16      (96 KB)
#define WS_PKHEAD  114688                    // 8*4*32*16 bf16       (32 KB)
#define WS_PKC1    147456                    // 8*4*32*16 bf16       (32 KB)
#define WS_PKC2    180224                    // 8*36*32*16 bf16     (288 KB)
#define WS_XWIN    524288                    // NWIN*64*128 bf16     (64 MB)  (reused as gwin)
#define WS_Y1WIN   (WS_XWIN + 67108864)      // NWIN*64*128 bf16     (64 MB)

// gfx1250 async global->LDS copy (ASYNCcnt path), guarded so either toolchain compiles.
#if defined(__HIP_DEVICE_COMPILE__) && \
    __has_builtin(__builtin_amdgcn_global_load_async_to_lds_b128) && \
    __has_builtin(__builtin_amdgcn_s_wait_asynccnt)
#define ASYNC_LDS 1
#else
#define ASYNC_LDS 0
#endif

__device__ __forceinline__ void cp16(void* lds, const void* g) {
#if ASYNC_LDS
  __builtin_amdgcn_global_load_async_to_lds_b128(
      (__attribute__((address_space(1))) av4i*)g,
      (__attribute__((address_space(3))) av4i*)lds, 0, 0);
#else
  *(float4*)lds = *(const float4*)g;
#endif
}
__device__ __forceinline__ void cp_wait() {
#if ASYNC_LDS
  __builtin_amdgcn_s_wait_asynccnt(0);
#endif
}

__device__ __forceinline__ float gelu_exact(float v) {
  return 0.5f * v * (1.0f + erff(v * 0.70710678118654752f));
}

// K offset (within a 32-wide k-slice) of fragment element e for this lane,
// per CDNA5 ISA 16-bit A/B VGPR layout (lanes 16-31 shifted by K+8; pairs 4..7 at K+16).
__device__ __forceinline__ int koff_of(int lane, int e) {
  int p = e >> 1;
  return ((p & 4) ? 16 : 0) + (p & 3) * 2 + ((lane >> 4) ? 8 : 0) + (e & 1);
}
__device__ __forceinline__ void make_koff(int lane, int* koff) {
#pragma unroll
  for (int e = 0; e < 16; ++e) koff[e] = koff_of(lane, e);
}

// ---------------- relative-position bias MLP (tiny) ----------------
__global__ void bias_kernel(const float* __restrict__ delta, const int* __restrict__ index,
                            const float* __restrict__ w_b1, const float* __restrict__ b_b1,
                            const float* __restrict__ w_b2, const float* __restrict__ b_b2,
                            float* __restrict__ bias, int U) {
  __shared__ float val[256];
  int t = threadIdx.x;
  for (int u = t; u < U; u += 256) {
    float d0 = delta[2 * u], d1 = delta[2 * u + 1];
    float acc = b_b2[0];
    for (int hd = 0; hd < 16; ++hd) {
      float hv = d0 * w_b1[2 * hd] + d1 * w_b1[2 * hd + 1] + b_b1[hd];
      acc += gelu_exact(hv) * w_b2[hd];
    }
    val[u] = acc;
  }
  __syncthreads();
  for (int i = t; i < NTOK * NTOK; i += 256) bias[i] = val[index[i]];
}

// -------- pack all weight matrices into per-lane WMMA B-fragment layout (bf16) --------
// packed[((nt*NKS + ks)*32 + lane)*16 + e] = bf16( B(ks*32+koff, nt*16+idx) )
__global__ void pack_weights(const float* __restrict__ w_qkv, const float* __restrict__ w_head,
                             const float* __restrict__ w_c1,  const float* __restrict__ w_c2,
                             __bf16* __restrict__ pq, __bf16* __restrict__ phd,
                             __bf16* __restrict__ p1, __bf16* __restrict__ p2) {
  int i = blockIdx.x * 256 + threadIdx.x;
  if (i < 49152) {                                  // qkv: NT=24, NKS=4
    int e = i & 15, lane = (i >> 4) & 31, t = i >> 9;
    int ks = t & 3, nt = t >> 2;
    int col = nt * 16 + (lane & 15), kk = ks * 32 + koff_of(lane, e);
    pq[i] = (__bf16)w_qkv[col * CCH + kk];
  } else if (i < 65536) {                           // head: NT=8, NKS=4
    int j = i - 49152;
    int e = j & 15, lane = (j >> 4) & 31, t = j >> 9;
    int ks = t & 3, nt = t >> 2;
    int col = nt * 16 + (lane & 15), kk = ks * 32 + koff_of(lane, e);
    phd[j] = (__bf16)w_head[col * CCH + kk];
  } else if (i < 81920) {                           // c1: NT=8, NKS=4
    int j = i - 65536;
    int e = j & 15, lane = (j >> 4) & 31, t = j >> 9;
    int ks = t & 3, nt = t >> 2;
    int col = nt * 16 + (lane & 15), kk = ks * 32 + koff_of(lane, e);
    p1[j] = (__bf16)w_c1[col * CCH + kk];
  } else if (i < 229376) {                          // c2: NT=8, NKS=36, K = tap*128 + ci
    int j = i - 81920;
    int e = j & 15, lane = (j >> 4) & 31, t = j >> 9;
    int ks = t % 36, nt = t / 36;
    int col = nt * 16 + (lane & 15), kk = ks * 32 + koff_of(lane, e);
    int tap = kk >> 7, ci = kk & 127;
    p2[j] = (__bf16)w_c2[((size_t)col * CCH + ci) * 9 + tap];
  }
}

// -------- pack x (fp32 NCHW) -> bf16 window-token layout xwin[win][n][c] --------
__global__ void pack_x(const float* __restrict__ x, __bf16* __restrict__ xwin) {
  for (size_t i = (size_t)blockIdx.x * 256 + threadIdx.x; i < (size_t)4 * CCH * HH * WW;
       i += (size_t)gridDim.x * 256) {
    int w = (int)(i & 255), h = (int)((i >> 8) & 255), c = (int)((i >> 16) & 127), b = (int)(i >> 23);
    int win = (b << 10) | ((h >> 3) << 5) | (w >> 3);
    int n = ((h & 7) << 3) | (w & 7);
    xwin[((size_t)win * NTOK + n) * CCH + c] = (__bf16)x[i];
  }
}

// ---------------- window attention + residual (y1 = x + attn_out) ----------------
__global__ void attn_kernel(const float* __restrict__ x, const __bf16* __restrict__ xwin,
                            const __bf16* __restrict__ pq, const float* __restrict__ b_qkv,
                            const __bf16* __restrict__ phd, const float* __restrict__ b_head,
                            const float* __restrict__ bias,
                            float* __restrict__ out, __bf16* __restrict__ y1win) {
  extern __shared__ char smem[];
  __bf16* xbf   = (__bf16*)smem;                       // 64*128 bf16 (reused as obf)
  __bf16* qkvbf = xbf + NTOK * CCH;                    // 64*384 bf16
  float*  sc    = (float*)(qkvbf + NTOK * 3 * CCH);    // 2*64*64 f32 (scores -> probs)
  __bf16* obf   = xbf;

  const int tid  = threadIdx.x;
  const int lane = tid & 31;
  const int wv   = tid >> 5;
  const int half = lane >> 4;
  const int idx  = lane & 15;
  int koff[16];
  make_koff(lane, koff);

  const int win = blockIdx.x;
  const int bb = win >> 10;
  const int h0 = ((win >> 5) & 31) * WSZ, w0 = (win & 31) * WSZ;

  // stage window: contiguous 16 KB bf16 copy (async global->LDS on gfx1250)
  {
    const char* src = (const char*)(xwin + (size_t)win * NTOK * CCH);
    for (int i = tid; i < 1024; i += 256) cp16((char*)xbf + i * 16, src + i * 16);
    cp_wait();
  }
  __syncthreads();

  // QKV: [64x128] @ [128x384]  (96 tiles, 4 k-steps); B frags pre-packed
  for (int t = wv; t < 96; t += 8) {
    int mt = t & 3, nt = t >> 2;
    v8f acc = {};
    for (int ks = 0; ks < 4; ++ks) {
      v16bf a, b;
#pragma unroll
      for (int e = 0; e < 16; ++e) a[e] = xbf[(mt * 16 + idx) * CCH + ks * 32 + koff[e]];
      b = *(const v16bf*)(pq + (size_t)(((nt * 4 + ks) * 32 + lane) << 4));
      acc = __builtin_amdgcn_wmma_f32_16x16x32_bf16(false, a, false, b, (short)0, acc, false, false);
    }
#pragma unroll
    for (int r = 0; r < 8; ++r) {
      int row = mt * 16 + half * 8 + r, col = nt * 16 + idx;
      qkvbf[row * (3 * CCH) + col] = (__bf16)(acc[r] + b_qkv[col]);
    }
  }
  __syncthreads();

  // scores = q k^T * 0.125 + bias
  for (int t = wv; t < 32; t += 8) {
    int hd = t >> 4, tt = t & 15, mt = tt >> 2, nt = tt & 3;
    v8f acc = {};
    for (int ks = 0; ks < 2; ++ks) {
      v16bf a, b;
#pragma unroll
      for (int e = 0; e < 16; ++e) {
        int kk = ks * 32 + koff[e];
        a[e] = qkvbf[(mt * 16 + idx) * (3 * CCH) + hd * DK + kk];
        b[e] = qkvbf[(nt * 16 + idx) * (3 * CCH) + CCH + hd * DK + kk];
      }
      acc = __builtin_amdgcn_wmma_f32_16x16x32_bf16(false, a, false, b, (short)0, acc, false, false);
    }
#pragma unroll
    for (int r = 0; r < 8; ++r) {
      int row = mt * 16 + half * 8 + r, col = nt * 16 + idx;
      sc[(hd * NTOK + row) * NTOK + col] = acc[r] * 0.125f + bias[row * NTOK + col];
    }
  }
  __syncthreads();

  // softmax per row; normalized probs kept in sc (f32)
  if (tid < NH * NTOK) {
    float* srow = sc + tid * NTOK;
    float m = srow[0];
    for (int i = 1; i < NTOK; ++i) m = fmaxf(m, srow[i]);
    float s = 0.f;
    for (int i = 0; i < NTOK; ++i) { float e = expf(srow[i] - m); srow[i] = e; s += e; }
    float inv = 1.0f / s;
    for (int i = 0; i < NTOK; ++i) srow[i] *= inv;
  }
  __syncthreads();

  // o = probs @ v ; o stored [n][h*64+d]
  for (int t = wv; t < 32; t += 8) {
    int hd = t >> 4, tt = t & 15, mt = tt >> 2, nt = tt & 3;
    v8f acc = {};
    for (int ks = 0; ks < 2; ++ks) {
      v16bf a, b;
#pragma unroll
      for (int e = 0; e < 16; ++e) {
        int kk = ks * 32 + koff[e];
        a[e] = (__bf16)sc[(hd * NTOK + mt * 16 + idx) * NTOK + kk];
        b[e] = qkvbf[kk * (3 * CCH) + 2 * CCH + hd * DK + nt * 16 + idx];
      }
      acc = __builtin_amdgcn_wmma_f32_16x16x32_bf16(false, a, false, b, (short)0, acc, false, false);
    }
#pragma unroll
    for (int r = 0; r < 8; ++r) {
      int row = mt * 16 + half * 8 + r, col = nt * 16 + idx;
      obf[row * CCH + hd * DK + col] = (__bf16)acc[r];
    }
  }
  __syncthreads();

  // head projection + residual: y1 = x + o @ w_head^T + b_head (fp32 out + bf16 window copy)
  for (int t = wv; t < 32; t += 8) {
    int mt = t >> 3, nt = t & 7;
    v8f acc = {};
    for (int ks = 0; ks < 4; ++ks) {
      v16bf a, b;
#pragma unroll
      for (int e = 0; e < 16; ++e) a[e] = obf[(mt * 16 + idx) * CCH + ks * 32 + koff[e]];
      b = *(const v16bf*)(phd + (size_t)(((nt * 4 + ks) * 32 + lane) << 4));
      acc = __builtin_amdgcn_wmma_f32_16x16x32_bf16(false, a, false, b, (short)0, acc, false, false);
    }
#pragma unroll
    for (int r = 0; r < 8; ++r) {
      int row = mt * 16 + half * 8 + r, col = nt * 16 + idx;
      size_t gi = (((size_t)bb * CCH + col) * HH + (h0 + (row >> 3))) * WW + (w0 + (row & 7));
      float v = x[gi] + acc[r] + b_head[col];
      out[gi] = v;
      y1win[((size_t)win * NTOK + row) * CCH + col] = (__bf16)v;
    }
  }
}

// ---------------- 1x1 conv + exact GELU -> gwin (bf16 window layout) ----------------
__global__ void conv1_kernel(const __bf16* __restrict__ y1win,
                             const __bf16* __restrict__ p1, const float* __restrict__ b_c1,
                             __bf16* __restrict__ gwin) {
  __shared__ __bf16 inbf[NTOK * CCH];
  const int tid = threadIdx.x, lane = tid & 31, wv = tid >> 5;
  const int half = lane >> 4, idx = lane & 15;
  int koff[16];
  make_koff(lane, koff);
  const int win = blockIdx.x;

  {
    const char* src = (const char*)(y1win + (size_t)win * NTOK * CCH);
    for (int i = tid; i < 1024; i += 256) cp16((char*)inbf + i * 16, src + i * 16);
    cp_wait();
  }
  __syncthreads();

  for (int t = wv; t < 32; t += 8) {
    int mt = t >> 3, nt = t & 7;
    v8f acc = {};
    for (int ks = 0; ks < 4; ++ks) {
      v16bf a, b;
#pragma unroll
      for (int e = 0; e < 16; ++e) a[e] = inbf[(mt * 16 + idx) * CCH + ks * 32 + koff[e]];
      b = *(const v16bf*)(p1 + (size_t)(((nt * 4 + ks) * 32 + lane) << 4));
      acc = __builtin_amdgcn_wmma_f32_16x16x32_bf16(false, a, false, b, (short)0, acc, false, false);
    }
#pragma unroll
    for (int r = 0; r < 8; ++r) {
      int row = mt * 16 + half * 8 + r, col = nt * 16 + idx;
      gwin[((size_t)win * NTOK + row) * CCH + col] = (__bf16)gelu_exact(acc[r] + b_c1[col]);
    }
  }
}

// ------- 3x3 conv (replicate pad) as implicit GEMM K=1152 + LeakyReLU + residual -------
__global__ void conv2_kernel(const __bf16* __restrict__ gwin,
                             const __bf16* __restrict__ p2, const float* __restrict__ b_c2,
                             float* __restrict__ out) {
  __shared__ __bf16 patch[10 * 10 * CCH];   // 25.6 KB clamped halo, [py*10+px][ci]
  const int tid = threadIdx.x, lane = tid & 31, wv = tid >> 5;
  const int half = lane >> 4, idx = lane & 15;
  int koff[16];
  make_koff(lane, koff);

  const int win = blockIdx.x;
  const int bb = win >> 10;
  const int h0 = ((win >> 5) & 31) * WSZ, w0 = (win & 31) * WSZ;

  // stage halo from bf16 window layout: 100 px * 256 B = 1600 16-byte chunks
  for (int i = tid; i < 1600; i += 256) {
    int pp = i >> 4, ch = i & 15;
    int py = pp / 10, px = pp - py * 10;
    int gh = h0 + py - 1; gh = gh < 0 ? 0 : (gh > HH - 1 ? HH - 1 : gh);
    int gw = w0 + px - 1; gw = gw < 0 ? 0 : (gw > WW - 1 ? WW - 1 : gw);
    int win2 = (bb << 10) | ((gh >> 3) << 5) | (gw >> 3);
    int n2 = ((gh & 7) << 3) | (gw & 7);
    cp16(patch + pp * CCH + ch * 8,
         gwin + ((size_t)win2 * NTOK + n2) * CCH + ch * 8);
  }
  cp_wait();
  __syncthreads();

  // K = 9 taps * 128 ch -> 36 k-steps of 32 (a step never crosses a tap)
  for (int t = wv; t < 32; t += 8) {
    int mt = t >> 3, nt = t & 7;
    int n = mt * 16 + idx, pr = n >> 3, pc = n & 7;
    v8f acc = {};
    for (int ks = 0; ks < 36; ++ks) {
      int tap = ks >> 2, cib = (ks & 3) * 32;
      int dy = tap / 3, dx = tap - dy * 3;
      const __bf16* arow = patch + ((pr + dy) * 10 + (pc + dx)) * CCH + cib;
      v16bf a, b;
#pragma unroll
      for (int e = 0; e < 16; ++e) a[e] = arow[koff[e]];
      b = *(const v16bf*)(p2 + (size_t)(((nt * 36 + ks) * 32 + lane) << 4));
      acc = __builtin_amdgcn_wmma_f32_16x16x32_bf16(false, a, false, b, (short)0, acc, false, false);
    }
#pragma unroll
    for (int r = 0; r < 8; ++r) {
      int row = mt * 16 + half * 8 + r, col = nt * 16 + idx;
      float v = acc[r] + b_c2[col];
      v = v > 0.f ? v : 0.1f * v;
      size_t gi = (((size_t)bb * CCH + col) * HH + (h0 + (row >> 3))) * WW + (w0 + (row & 7));
      out[gi] = out[gi] + v;   // out holds y1; final = y1 + leaky(conv2)
    }
  }
}

extern "C" void kernel_launch(void* const* d_in, const int* in_sizes, int n_in,
                              void* d_out, int out_size, void* d_ws, size_t ws_size,
                              hipStream_t stream) {
  const float* x      = (const float*)d_in[0];
  const float* delta  = (const float*)d_in[1];
  const int*   index  = (const int*)d_in[2];
  const float* w_qkv  = (const float*)d_in[3];
  const float* b_qkv  = (const float*)d_in[4];
  const float* w_head = (const float*)d_in[5];
  const float* b_head = (const float*)d_in[6];
  const float* w_b1   = (const float*)d_in[7];
  const float* b_b1   = (const float*)d_in[8];
  const float* w_b2   = (const float*)d_in[9];
  const float* b_b2   = (const float*)d_in[10];
  const float* w_c1   = (const float*)d_in[11];
  const float* b_c1   = (const float*)d_in[12];
  const float* w_c2   = (const float*)d_in[13];
  const float* b_c2   = (const float*)d_in[14];
  float* out = (float*)d_out;

  char* ws = (char*)d_ws;
  float*  bias  = (float*)(ws + WS_BIAS);
  __bf16* pq    = (__bf16*)(ws + WS_PKQKV);
  __bf16* phd   = (__bf16*)(ws + WS_PKHEAD);
  __bf16* p1    = (__bf16*)(ws + WS_PKC1);
  __bf16* p2    = (__bf16*)(ws + WS_PKC2);
  __bf16* xwin  = (__bf16*)(ws + WS_XWIN);
  __bf16* y1win = (__bf16*)(ws + WS_Y1WIN);
  __bf16* gwin  = xwin;   // xwin dead after attn; reuse for gelu buffer
  int U = in_sizes[1] / 2;

  size_t attn_lds = (size_t)(NTOK * CCH + NTOK * 3 * CCH) * sizeof(__bf16)
                  + (size_t)NH * NTOK * NTOK * sizeof(float);   // 96 KB < 320 KB WGP LDS
  (void)hipFuncSetAttribute((const void*)attn_kernel,
                            hipFuncAttributeMaxDynamicSharedMemorySize, (int)attn_lds);

  bias_kernel  <<<1,    256, 0,        stream>>>(delta, index, w_b1, b_b1, w_b2, b_b2, bias, U);
  pack_weights <<<896,  256, 0,        stream>>>(w_qkv, w_head, w_c1, w_c2, pq, phd, p1, p2);
  pack_x       <<<8192, 256, 0,        stream>>>(x, xwin);
  attn_kernel  <<<NWIN, 256, attn_lds, stream>>>(x, xwin, pq, b_qkv, phd, b_head, bias, out, y1win);
  conv1_kernel <<<NWIN, 256, 0,        stream>>>(y1win, p1, b_c1, gwin);
  conv2_kernel <<<NWIN, 256, 0,        stream>>>(gwin, p2, b_c2, out);
}
